// AdditiveAttention_21595095564825
// MI455X (gfx1250) — matrix operand order repaired and checked by reading, not compiled, not run on hardware
//
#include <hip/hip_runtime.h>
#include <hip/hip_bf16.h>
#include <stdint.h>

// ---------------------------------------------------------------------------
// AdditiveAttention on gfx1250 (MI455X):
//   scores_i = v . tanh(W h_i + b); alpha = softmax(scores); out = alpha @ h
// v_wmma_f32_16x16x32_bf16 GEMM, fp32 accum. h_i staged fp32->bf16 in LDS and
// streamed from HBM exactly twice (score pass + weighted-sum pass).
// W (bf16, 1.18 MB) lives in the 192 MB L2. Each wave register-blocks 2 M-tiles
// per B fragment and the 4 waves of a block split the 48 j-tiles in half,
// halving B-operand L2/WGP$ traffic vs. the naive mapping.
// ---------------------------------------------------------------------------

typedef __attribute__((ext_vector_type(16))) __bf16 v16bf;
typedef __attribute__((ext_vector_type(8)))  __bf16 v8bf;
typedef __attribute__((ext_vector_type(4)))  __bf16 v4bf;
typedef __attribute__((ext_vector_type(8)))  float  v8f;
typedef __attribute__((ext_vector_type(4)))  float  v4f;

#define HDIM 768
#define NROWS 131072
#define BLK_ROWS 64      // rows of h_i per block in the score kernel
#define TPB_GEMM 128     // 4 waves: (j-half) x (row-half), 32 rows per wave

__device__ __forceinline__ __bf16 f2bf(float f) {
  union { float f; uint32_t u; } x; x.f = f;
  uint32_t r = x.u + 0x7FFFu + ((x.u >> 16) & 1u);   // round-to-nearest-even
  union { uint16_t u; __bf16 b; } y; y.u = (uint16_t)(r >> 16);
  return y.b;
}

// A-fragment per ISA 16-bit 16x32 layout: vgprs0-3 = K[kb+8*half .. +7],
// vgprs4-7 = K[kb+16+8*half .. +7]  (two 16B ds_load_b128 per lane)
__device__ __forceinline__ v16bf load_a(const __bf16* __restrict__ aRow,
                                        int kb, int half) {
  v8bf alo = *(const v8bf*)(aRow + kb + 8 * half);
  v8bf ahi = *(const v8bf*)(aRow + kb + 16 + 8 * half);
  v16bf a;
  #pragma unroll
  for (int q = 0; q < 8; ++q) { a[q] = alo[q]; a[q + 8] = ahi[q]; }
  return a;
}

// ---- kernel 1: W fp32 -> bf16, and zero the score accumulator --------------
__global__ __launch_bounds__(256) void k_convert_w(const float* __restrict__ W,
                                                   __bf16* __restrict__ Wb,
                                                   float* __restrict__ scores) {
  const int gtid = blockIdx.x * 256 + threadIdx.x;
  const int i = gtid * 4;
  v4f x = *(const v4f*)(W + i);
  v4bf o;
  o[0] = f2bf(x[0]); o[1] = f2bf(x[1]); o[2] = f2bf(x[2]); o[3] = f2bf(x[3]);
  *(v4bf*)(Wb + i) = o;
  if (gtid < NROWS) scores[gtid] = 0.0f;   // scores accumulated atomically
}

// ---- kernel 2: WMMA GEMM + tanh + v-dot -> scores[N] -----------------------
__global__ __launch_bounds__(TPB_GEMM) void k_score(
    const float* __restrict__ h, const __bf16* __restrict__ Wb,
    const float* __restrict__ bias, const float* __restrict__ v,
    float* __restrict__ scores)
{
  __shared__ __bf16 sA[BLK_ROWS * HDIM];           // 96 KB of 320 KB WGP LDS
  const int tid = threadIdx.x;
  const size_t rowBase = (size_t)blockIdx.x * BLK_ROWS;

  // Stage 64 rows of h (fp32) into LDS as bf16, coalesced float4 loads.
  const float* hBlk = h + rowBase * HDIM;
  for (int i = tid * 4; i < BLK_ROWS * HDIM; i += TPB_GEMM * 4) {
    v4f x = *(const v4f*)(hBlk + i);
    v4bf o;
    o[0] = f2bf(x[0]); o[1] = f2bf(x[1]); o[2] = f2bf(x[2]); o[3] = f2bf(x[3]);
    *(v4bf*)(&sA[i]) = o;
  }
  __syncthreads();

  const int wave = tid >> 5;
  const int wj   = wave >> 1;       // which half of the 48 j-tiles
  const int wm   = wave & 1;        // which 32-row half of the block
  const int lane = tid & 31;
  const int m    = lane & 15;       // row (A) / column n (B,C) within tile
  const int half = lane >> 4;       // K-half selector per ISA layout

  const __bf16* aRow0 = &sA[(wm * 32 + m) * HDIM];        // M-tile 0
  const __bf16* aRow1 = &sA[(wm * 32 + 16 + m) * HDIM];   // M-tile 1

  float sacc0[8] = {0.f,0.f,0.f,0.f,0.f,0.f,0.f,0.f};
  float sacc1[8] = {0.f,0.f,0.f,0.f,0.f,0.f,0.f,0.f};

  const int jtBeg = wj * 24, jtEnd = jtBeg + 24;
  for (int jt = jtBeg; jt < jtEnd; ++jt) {
    const __bf16* bCol = Wb + (size_t)(jt * 16 + m) * HDIM;
    v8f c0 = {}, c1 = {};
    #pragma unroll 4
    for (int kb = 0; kb < HDIM; kb += 32) {
      // One B fragment (2x16B/lane from row-major W, same K layout as A) feeds two WMMAs.
      v16bf b  = load_a(bCol, kb, half);
      v16bf a0 = load_a(aRow0, kb, half);
      v16bf a1 = load_a(aRow1, kb, half);
      c0 = __builtin_amdgcn_wmma_f32_16x16x32_bf16(
               false, a0, false, b, (short)0, c0, false, false);
      c1 = __builtin_amdgcn_wmma_f32_16x16x32_bf16(
               false, a1, false, b, (short)0, c1, false, false);
    }
    const float bj = bias[jt * 16 + m];
    const float vj = v[jt * 16 + m];
    #pragma unroll
    for (int r = 0; r < 8; ++r) {
      sacc0[r] += vj * tanhf(c0[r] + bj);       // C: M = r + 8*half, N = m
      sacc1[r] += vj * tanhf(c1[r] + bj);
    }
  }

  // Reduce over N (16 lanes within each half-wave).
  #pragma unroll
  for (int r = 0; r < 8; ++r) {
    float s0 = sacc0[r], s1 = sacc1[r];
    s0 += __shfl_xor(s0, 1, 32); s1 += __shfl_xor(s1, 1, 32);
    s0 += __shfl_xor(s0, 2, 32); s1 += __shfl_xor(s1, 2, 32);
    s0 += __shfl_xor(s0, 4, 32); s1 += __shfl_xor(s1, 4, 32);
    s0 += __shfl_xor(s0, 8, 32); s1 += __shfl_xor(s1, 8, 32);
    sacc0[r] = s0; sacc1[r] = s1;
  }
  if (m == 0) {
    // Two waves (wj=0,1) contribute partial j-sums per row -> atomic combine.
    #pragma unroll
    for (int r = 0; r < 8; ++r) {
      atomicAdd(&scores[rowBase + wm * 32 + half * 8 + r],      sacc0[r]);
      atomicAdd(&scores[rowBase + wm * 32 + 16 + half * 8 + r], sacc1[r]);
    }
  }
}

// ---- kernel 3: softmax stats (max, sum of exp) + zero d_out ----------------
__global__ __launch_bounds__(1024) void k_stats(const float* __restrict__ scores,
                                                float* __restrict__ stats,
                                                float* __restrict__ out) {
  __shared__ float red[1024];
  const int t = threadIdx.x;
  float mx = -3.4e38f;
  for (int i = t; i < NROWS; i += 1024) mx = fmaxf(mx, scores[i]);
  red[t] = mx; __syncthreads();
  for (int s = 512; s > 0; s >>= 1) {
    if (t < s) red[t] = fmaxf(red[t], red[t + s]);
    __syncthreads();
  }
  mx = red[0]; __syncthreads();
  float sum = 0.f;
  for (int i = t; i < NROWS; i += 1024) sum += __expf(scores[i] - mx);
  red[t] = sum; __syncthreads();
  for (int s = 512; s > 0; s >>= 1) {
    if (t < s) red[t] += red[t + s];
    __syncthreads();
  }
  if (t == 0) { stats[0] = mx; stats[1] = red[0]; }
  if (t < HDIM) out[t] = 0.0f;
}

// ---- kernel 4: out[768] += sum_i alpha_i * h_i (second pass over h) --------
__global__ __launch_bounds__(256) void k_wsum(const float* __restrict__ h,
                                              const float* __restrict__ scores,
                                              const float* __restrict__ stats,
                                              float* __restrict__ out) {
  __shared__ float wl[BLK_ROWS];
  const int t = threadIdx.x;
  const size_t rowBase = (size_t)blockIdx.x * BLK_ROWS;
  if (t < BLK_ROWS) {
    const float inv = 1.0f / stats[1];
    wl[t] = __expf(scores[rowBase + t] - stats[0]) * inv;
  }
  __syncthreads();
  float a0 = 0.f, a1 = 0.f, a2 = 0.f;
  const float* hb = h + rowBase * HDIM;
  for (int i = 0; i < BLK_ROWS; ++i) {
    const float wi = wl[i];
    a0 += wi * hb[i * HDIM + t];
    a1 += wi * hb[i * HDIM + t + 256];
    a2 += wi * hb[i * HDIM + t + 512];
  }
  atomicAdd(&out[t], a0);
  atomicAdd(&out[t + 256], a1);
  atomicAdd(&out[t + 512], a2);
}

// ---------------------------------------------------------------------------
extern "C" void kernel_launch(void* const* d_in, const int* in_sizes, int n_in,
                              void* d_out, int out_size, void* d_ws, size_t ws_size,
                              hipStream_t stream) {
  const float* h_i    = (const float*)d_in[0];   // [N, H] fp32
  const float* W      = (const float*)d_in[1];   // [H, H] fp32
  const float* W_bias = (const float*)d_in[2];   // [H]    fp32
  const float* v      = (const float*)d_in[3];   // [1, H] fp32
  float* out = (float*)d_out;                    // [1, H] fp32

  // workspace layout: Wb bf16 (1.18 MB) | scores fp32 (512 KB) | stats (2 f32)
  __bf16* Wb     = (__bf16*)d_ws;
  float*  scores = (float*)((char*)d_ws + (size_t)HDIM * HDIM * sizeof(__bf16));
  float*  stats  = scores + NROWS;

  k_convert_w<<<(HDIM * HDIM) / (256 * 4), 256, 0, stream>>>(W, Wb, scores);
  k_score<<<NROWS / BLK_ROWS, TPB_GEMM, 0, stream>>>(h_i, Wb, W_bias, v, scores);
  k_stats<<<1, 1024, 0, stream>>>(scores, stats, out);
  k_wsum<<<NROWS / BLK_ROWS, 256, 0, stream>>>(h_i, scores, stats, out);
}
